// GIN_14053132992702
// MI455X (gfx1250) — matrix operand order repaired and checked by reading, not compiled, not run on hardware
//
#include <hip/hip_runtime.h>

// GIN forward on MI455X (gfx1250, wave32).
// - Aggregation: global_atomic_add_f32 scatter (L2-resident, 192MB L2 holds all buffers)
// - GEMMs: v_wmma_f32_16x16x32_bf16 (bf16 operands, f32 accumulate)
// - BN stats fused into elementwise passes where possible.

#define GIN_N 100000
#define GIN_E 640000
#define GIN_D 128
#define GIN_L 4
#define BN_EPS 1e-5f

typedef __attribute__((ext_vector_type(8)))  float  v8f;
typedef __attribute__((ext_vector_type(16))) __bf16 v16bf;

// ---------------------------------------------------------------- small utils

__global__ __launch_bounds__(256)
void zero_kernel(float* __restrict__ p, int n) {
    int i = blockIdx.x * blockDim.x + threadIdx.x;
    if (i < n) p[i] = 0.0f;
}

__global__ __launch_bounds__(256)
void copy4_kernel(float4* __restrict__ dst, const float4* __restrict__ src, int n4) {
    int i = blockIdx.x * blockDim.x + threadIdx.x;
    if (i < n4) dst[i] = src[i];
}

// ------------------------------------------------ weight packing (f32 -> bf16)
// Pack W[k][n] (row-major DxD) into per-lane-contiguous WMMA B fragments:
// frag index = ((mat*4 + ks)*8 + ntile)*512 + lane*16 + j
// lane: col n = ntile*16 + (lane&15); K = ks*32 + ((lane>>4)&1)*8 + (j&7) + (j>>3)*16

__global__ __launch_bounds__(256)
void pack_w_kernel(__bf16* __restrict__ Bp,
                   const float* __restrict__ W1, const float* __restrict__ W2) {
    int idx = blockIdx.x * blockDim.x + threadIdx.x;
    const int total = 2 * GIN_L * GIN_D * GIN_D;
    if (idx >= total) return;
    int mat  = idx / 16384;          // layer*2 + which
    int rem  = idx % 16384;
    int ks   = rem / 4096;
    int rem2 = rem % 4096;
    int t    = rem2 / 512;
    int rem3 = rem2 % 512;
    int lane = rem3 / 16;
    int j    = rem3 % 16;
    int n = t * 16 + (lane & 15);
    int k = ks * 32 + ((lane >> 4) & 1) * 8 + (j & 7) + (j >> 3) * 16;
    int layer = mat >> 1;
    const float* W = (mat & 1) ? (W2 + layer * GIN_D * GIN_D)
                               : (W1 + layer * GIN_D * GIN_D);
    Bp[idx] = (__bf16)W[k * GIN_D + n];
}

// ------------------------------------------------------- edge scatter (atomics)
// R[dst] += X[src]  (float4 gather, 4 fp32 global atomics per thread)

__global__ __launch_bounds__(256)
void edge_kernel(float* __restrict__ R, const float* __restrict__ X,
                 const int* __restrict__ esrc, const int* __restrict__ edst,
                 int nEdges) {
    int tid = blockIdx.x * blockDim.x + threadIdx.x;
    int e = tid >> 5;
    if (e >= nEdges) return;
    int c = (tid & 31) << 2;
    int s = esrc[e], d = edst[e];
    const float4 v = *(const float4*)(X + (size_t)s * GIN_D + c);
    float* p = R + (size_t)d * GIN_D + c;
    unsafeAtomicAdd(p + 0, v.x);
    unsafeAtomicAdd(p + 1, v.y);
    unsafeAtomicAdd(p + 2, v.z);
    unsafeAtomicAdd(p + 3, v.w);
}

// ------------------------------------------------------------- WMMA GEMM + bias
// T[N x 128] = A[N x 128] @ B[128 x 128] + bias.
// 256 threads = 8 waves/block; each wave: 16 rows x 128 cols (8 accum tiles).
// A converted f32->bf16 in-register; B pre-packed bf16 fragments.

__global__ __launch_bounds__(256)
void gemm_bias_kernel(float* __restrict__ Tout, const float* __restrict__ A,
                      const __bf16* __restrict__ Bp, const float* __restrict__ bias,
                      int nRows) {
    int wave = threadIdx.x >> 5;
    int lane = threadIdx.x & 31;
    int rowbase = (blockIdx.x * 8 + wave) * 16;
    int m  = lane & 15;
    int hi = (lane >> 4) & 1;

    int lrow = rowbase + m;
    if (lrow >= nRows) lrow = nRows - 1;      // clamp loads; stores predicated
    const float* arow = A + (size_t)lrow * GIN_D;

    v8f acc[8] = {};
    #pragma unroll
    for (int ks = 0; ks < 4; ++ks) {
        int klo = ks * 32 + hi * 8;
        // ISA 16-bit A layout: lane<16 -> K klo..klo+7 & klo+16..klo+23 (klo base 0)
        //                      lane>=16 -> same with +8 offset (hi=1)
        v8f a0 = *(const v8f*)(arow + klo);
        v8f a1 = *(const v8f*)(arow + klo + 16);
        v16bf afrag;
        #pragma unroll
        for (int j = 0; j < 8; ++j) {
            afrag[j]     = (__bf16)a0[j];
            afrag[8 + j] = (__bf16)a1[j];
        }
        const __bf16* bbase = Bp + (size_t)(ks * 8) * 512 + lane * 16;
        #pragma unroll
        for (int t = 0; t < 8; ++t) {
            v16bf bfrag = *(const v16bf*)(bbase + (size_t)t * 512);
            acc[t] = __builtin_amdgcn_wmma_f32_16x16x32_bf16(
                false, afrag, false, bfrag, (short)0, acc[t], false, false);
        }
    }
    #pragma unroll
    for (int t = 0; t < 8; ++t) {
        int col = t * 16 + m;
        float bv = bias[col];
        #pragma unroll
        for (int g = 0; g < 8; ++g) {
            int r = rowbase + g + hi * 8;     // C/D layout: VGPR g -> row g (+8 hi half)
            if (r < nRows) Tout[(size_t)r * GIN_D + col] = acc[t][g] + bv;
        }
    }
}

// -------------------------------------------------------- column stats / pooling
// 128 threads/block, thread d owns column d over a row slab; atomics finalize.

__global__ __launch_bounds__(128)
void colstats_kernel(float* __restrict__ stats, const float* __restrict__ X,
                     int nRows, int rowsPerBlock) {
    int d = threadIdx.x;
    int r0 = blockIdx.x * rowsPerBlock;
    int r1 = min(r0 + rowsPerBlock, nRows);
    float s = 0.f, sq = 0.f;
    for (int r = r0; r < r1; ++r) {
        float v = X[(size_t)r * GIN_D + d];
        s += v; sq += v * v;
    }
    unsafeAtomicAdd(&stats[d], s);
    unsafeAtomicAdd(&stats[GIN_D + d], sq);
}

__global__ __launch_bounds__(128)
void colsum_kernel(float* __restrict__ pool, const float* __restrict__ X,
                   int nRows, int rowsPerBlock) {
    int d = threadIdx.x;
    int r0 = blockIdx.x * rowsPerBlock;
    int r1 = min(r0 + rowsPerBlock, nRows);
    float s = 0.f;
    for (int r = r0; r < r1; ++r) s += X[(size_t)r * GIN_D + d];
    unsafeAtomicAdd(&pool[d], s);
}

// --------------------------------------------- fused BN (training stats) + ReLU
// Y = relu((X-mean)*rsqrt(var+eps)*g + be); optionally accumulates post-ReLU
// stats (for the next BN) and/or the sum-pool for readout.

__global__ __launch_bounds__(128)
void bnrelu_kernel(float* __restrict__ Y, const float* __restrict__ X,
                   const float* __restrict__ stats,
                   const float* __restrict__ g, const float* __restrict__ be,
                   float* __restrict__ stats_out, float* __restrict__ pool_out,
                   int nRows, int rowsPerBlock, float invN) {
    int d = threadIdx.x;
    float mean = stats[d] * invN;
    float var  = stats[GIN_D + d] * invN - mean * mean;
    float scale = rsqrtf(var + BN_EPS) * g[d];
    float shift = be[d] - mean * scale;
    int r0 = blockIdx.x * rowsPerBlock;
    int r1 = min(r0 + rowsPerBlock, nRows);
    float s = 0.f, sq = 0.f;
    for (int r = r0; r < r1; ++r) {
        float v = X[(size_t)r * GIN_D + d] * scale + shift;
        v = fmaxf(v, 0.f);
        Y[(size_t)r * GIN_D + d] = v;
        s += v; sq += v * v;
    }
    if (stats_out) {
        unsafeAtomicAdd(&stats_out[d], s);
        unsafeAtomicAdd(&stats_out[GIN_D + d], sq);
    }
    if (pool_out) unsafeAtomicAdd(&pool_out[d], s);
}

// --------------------------------------------------------------------- readout
// score = sum_i pooled_i . Wp[i] + sum_i bp[i]   (OUT == 1)

__global__ __launch_bounds__(128)
void readout_kernel(float* __restrict__ out, const float* __restrict__ pooled,
                    const float* __restrict__ Wp, const float* __restrict__ bp) {
    __shared__ float sdata[GIN_D];
    int d = threadIdx.x;
    float acc = 0.f;
    #pragma unroll
    for (int i = 0; i < GIN_L + 1; ++i)
        acc += pooled[i * GIN_D + d] * Wp[i * GIN_D + d];
    sdata[d] = acc;
    __syncthreads();
    for (int off = 64; off > 0; off >>= 1) {
        if (d < off) sdata[d] += sdata[d + off];
        __syncthreads();
    }
    if (d == 0) {
        float b = 0.f;
        #pragma unroll
        for (int i = 0; i < GIN_L + 1; ++i) b += bp[i];
        out[0] = sdata[0] + b;
    }
}

// ----------------------------------------------------------------------- driver

extern "C" void kernel_launch(void* const* d_in, const int* in_sizes, int n_in,
                              void* d_out, int out_size, void* d_ws, size_t ws_size,
                              hipStream_t stream) {
    const float* h   = (const float*)d_in[0];
    const int*   src = (const int*)  d_in[1];
    const int*   dst = (const int*)  d_in[2];
    const float* W1  = (const float*)d_in[3];
    const float* b1  = (const float*)d_in[4];
    const float* g1  = (const float*)d_in[5];
    const float* be1 = (const float*)d_in[6];
    const float* W2  = (const float*)d_in[7];
    const float* b2  = (const float*)d_in[8];
    const float* g2  = (const float*)d_in[9];
    const float* be2 = (const float*)d_in[10];
    const float* g3  = (const float*)d_in[11];
    const float* be3 = (const float*)d_in[12];
    const float* Wp  = (const float*)d_in[13];
    const float* bp  = (const float*)d_in[14];

    // workspace layout
    char*  ws     = (char*)d_ws;
    float* pooled = (float*)ws;                         // (L+1)*128 = 640 f
    float* stats  = pooled + (GIN_L + 1) * GIN_D;       // L*3 pairs * 256 = 3072 f
    const int nSmall = (GIN_L + 1) * GIN_D + GIN_L * 3 * 2 * GIN_D;
    __bf16* Bp = (__bf16*)(ws + 16384);                 // 2*L*128*128 bf16 = 256 KiB
    const size_t ND = (size_t)GIN_N * GIN_D;
    float* X = (float*)(ws + 16384 + 2 * GIN_L * GIN_D * GIN_D * sizeof(__bf16));
    float* R = X + ND;
    float* T = R + ND;

    const float invN = 1.0f / (float)GIN_N;
    const int RPB = 256;                                // rows per stats block
    const int statBlocks = (GIN_N + RPB - 1) / RPB;     // 391
    const int gemmBlocks = (GIN_N + 127) / 128;         // 782
    const int copyBlocks = (int)(ND / 4 / 256);         // 12500
    const int edgeBlocks = GIN_E * 32 / 256;            // 80000

    // init small region (harness poisons ws; must re-zero every call)
    zero_kernel<<<(nSmall + 255) / 256, 256, 0, stream>>>((float*)ws, nSmall);
    // pack weights to bf16 WMMA-B fragments
    pack_w_kernel<<<(2 * GIN_L * GIN_D * GIN_D + 255) / 256, 256, 0, stream>>>(Bp, W1, W2);
    // pooled[0] = colsum(h)
    colsum_kernel<<<statBlocks, GIN_D, 0, stream>>>(pooled, h, GIN_N, RPB);

    for (int l = 0; l < GIN_L; ++l) {
        const float* xin = (l == 0) ? h : X;
        float* sA = stats + (l * 3 + 0) * 2 * GIN_D;
        float* sB = stats + (l * 3 + 1) * 2 * GIN_D;
        float* sC = stats + (l * 3 + 2) * 2 * GIN_D;

        // r = x + segment_sum(x[src], dst)
        copy4_kernel<<<copyBlocks, 256, 0, stream>>>((float4*)R, (const float4*)xin,
                                                     (int)(ND / 4));
        edge_kernel<<<edgeBlocks, 256, 0, stream>>>(R, xin, src, dst, GIN_E);

        // t1 = r @ W1 + b1  (WMMA bf16)
        gemm_bias_kernel<<<gemmBlocks, 256, 0, stream>>>(
            T, R, Bp + (size_t)(l * 2 + 0) * 16384, b1 + l * GIN_D, GIN_N);
        colstats_kernel<<<statBlocks, GIN_D, 0, stream>>>(sA, T, GIN_N, RPB);
        // z1 = relu(bn(t1))  -> X (x is dead)
        bnrelu_kernel<<<statBlocks, GIN_D, 0, stream>>>(
            X, T, sA, g1 + l * GIN_D, be1 + l * GIN_D, nullptr, nullptr,
            GIN_N, RPB, invN);

        // t2 = z1 @ W2 + b2  -> R
        gemm_bias_kernel<<<gemmBlocks, 256, 0, stream>>>(
            R, X, Bp + (size_t)(l * 2 + 1) * 16384, b2 + l * GIN_D, GIN_N);
        colstats_kernel<<<statBlocks, GIN_D, 0, stream>>>(sB, R, GIN_N, RPB);
        // z2 = relu(bn(t2)) -> T, fused accumulation of post-ReLU stats for bn3
        bnrelu_kernel<<<statBlocks, GIN_D, 0, stream>>>(
            T, R, sB, g2 + l * GIN_D, be2 + l * GIN_D, sC, nullptr,
            GIN_N, RPB, invN);
        // x_{l+1} = relu(bn(z2)) -> X, fused sum-pool
        bnrelu_kernel<<<statBlocks, GIN_D, 0, stream>>>(
            X, T, sC, g3 + l * GIN_D, be3 + l * GIN_D, nullptr,
            pooled + (l + 1) * GIN_D, GIN_N, RPB, invN);
    }

    readout_kernel<<<1, GIN_D, 0, stream>>>((float*)d_out, pooled, Wp, bp);
    (void)in_sizes; (void)n_in; (void)out_size; (void)ws_size;
}
